// MoE_61838939128385
// MI455X (gfx1250) — matrix operand order repaired and checked by reading, not compiled, not run on hardware
//
#include <hip/hip_runtime.h>
#include <hip/hip_bf16.h>

typedef __attribute__((ext_vector_type(16))) __bf16 v16bf;
typedef __attribute__((ext_vector_type(8)))  __bf16 v8bf;
typedef __attribute__((ext_vector_type(8)))  float  v8f;

#define TOK   8192      // B*S
#define DDIM  1024
#define HDIM  2048
#define ODIM  1024
#define NEXP  8
#define KT    64        // K tile per staging round (2 WMMA substeps)
#define KP    72        // LDS K pitch in bf16 elems (64 data + 8 pad; 144B = 9*16B)
#define ROWS_MAX (2*TOK + NEXP*128)

union FragBF { v16bf v; v8bf h[2]; };

// A fragment (16x32 bf16, MxK) at K offset kk. ISA 7.12.2: lanes 0-15 hold
// M=lane, VGPR0-3 = K 0..7, VGPR4-7 = K 16..23; lanes 16-31 shift K by +8.
__device__ inline v16bf ld_fragA(const __bf16* base, int m, int half, int kk) {
  FragBF f;
  const __bf16* p = base + m * KP + kk + half * 8;
  f.h[0] = *(const v8bf*)(p);
  f.h[1] = *(const v8bf*)(p + 16);
  return f.v;
}
// B fragment (32x16 bf16, KxN) from LDS stored transposed as Bt[n][k].
// lanes 0-15: N=lane, element i<8 -> K=half*8+i, i>=8 -> K=16+half*8+(i-8).
__device__ inline v16bf ld_fragB(const __bf16* base, int n, int half, int kk) {
  FragBF f;
  const __bf16* p = base + n * KP + kk + half * 8;
  f.h[0] = *(const v8bf*)(p);
  f.h[1] = *(const v8bf*)(p + 16);
  return f.v;
}
__device__ inline v8bf cvt8(float4 a, float4 b) {
  v8bf r;
  r[0]=(__bf16)a.x; r[1]=(__bf16)a.y; r[2]=(__bf16)a.z; r[3]=(__bf16)a.w;
  r[4]=(__bf16)b.x; r[5]=(__bf16)b.y; r[6]=(__bf16)b.z; r[7]=(__bf16)b.w;
  return r;
}

// ---------------- gate: logits -> softmax -> top-2 + per-expert counts ----
__global__ __launch_bounds__(256) void moe_gate(
    const float* __restrict__ x, const float* __restrict__ gw,
    const float* __restrict__ gb, int* __restrict__ counts,
    int* __restrict__ tk_e, int* __restrict__ tk_p, float* __restrict__ tk_v)
{
  const int t = blockIdx.x;
  const int tid = threadIdx.x;
  const int lane = tid & 31, e = tid >> 5;      // 8 waves = 8 experts
  const float* xr = x + (size_t)t * DDIM;
  const float* wr = gw + (size_t)e * DDIM;
  float s = 0.0f;
  for (int d = lane; d < DDIM; d += 32) s += xr[d] * wr[d];
  for (int off = 16; off; off >>= 1) s += __shfl_xor(s, off, 32);
  __shared__ float lg[NEXP];
  if (lane == 0) lg[e] = s + gb[e];
  __syncthreads();
  if (tid == 0) {
    float m = lg[0];
    for (int i = 1; i < NEXP; i++) m = fmaxf(m, lg[i]);
    float p[NEXP], den = 0.0f;
    for (int i = 0; i < NEXP; i++) { p[i] = expf(lg[i] - m); den += p[i]; }
    const float inv = 1.0f / den;
    for (int i = 0; i < NEXP; i++) p[i] *= inv;
    int e0 = 0;
    for (int i = 1; i < NEXP; i++) if (p[i] > p[e0]) e0 = i;
    int e1 = (e0 == 0) ? 1 : 0;
    for (int i = 0; i < NEXP; i++) if (i != e0 && p[i] > p[e1]) e1 = i;
    tk_e[t*2+0] = e0; tk_v[t*2+0] = p[e0]; tk_p[t*2+0] = atomicAdd(&counts[e0], 1);
    tk_e[t*2+1] = e1; tk_v[t*2+1] = p[e1]; tk_p[t*2+1] = atomicAdd(&counts[e1], 1);
  }
}

// ---------------- tiny prefix scan of 128-padded per-expert offsets -------
__global__ void moe_offsets(const int* __restrict__ counts, int* __restrict__ offsets)
{
  if (threadIdx.x == 0 && blockIdx.x == 0) {
    int o = 0;
    for (int e = 0; e < NEXP; e++) { offsets[e] = o; o += ((counts[e] + 127) >> 7) << 7; }
    offsets[NEXP] = o;
  }
}

// ---------------- scatter (token,slot) -> compact expert row lists --------
__global__ __launch_bounds__(256) void moe_scatter(
    const int* __restrict__ offsets, const int* __restrict__ tk_e,
    const int* __restrict__ tk_p, const float* __restrict__ tk_v,
    int* __restrict__ row_token, float* __restrict__ row_gate)
{
  const int i = blockIdx.x * 256 + threadIdx.x;   // < 2*TOK
  const int t = i >> 1;
  const int e = tk_e[i];
  const int r = offsets[e] + tk_p[i];
  row_token[r] = t;
  row_gate[r]  = tk_v[i];
}

// ---------------- GEMM1: h = relu(gather(x) @ w1[e] + b1[e]) -> bf16 ws ---
__global__ __launch_bounds__(256) void moe_gemm1(
    const float* __restrict__ x, const float* __restrict__ w1,
    const float* __restrict__ b1, const int* __restrict__ counts,
    const int* __restrict__ offsets, const int* __restrict__ row_token,
    __bf16* __restrict__ hws)
{
  __shared__ __align__(16) __bf16 As[128 * KP];
  __shared__ __align__(16) __bf16 Bs[128 * KP];
  const int e = blockIdx.z;
  if ((int)blockIdx.y * 128 >= counts[e]) return;
  const int r0 = offsets[e] + blockIdx.y * 128;
  const int n0 = blockIdx.x * 128;

  const int tid  = threadIdx.x;
  const int lane = tid & 31, wave = tid >> 5;
  const int half = lane >> 4, lm = lane & 15;
  const int mblk = (wave & 3) * 32;
  const int nblk = (wave >> 2) * 64;

  // A-staging role: 2 threads/row, 32 cols each
  const int arow = tid >> 1;
  const int acol = (tid & 1) * 32;
  const int tokA = row_token[r0 + arow];
  const float* xrow = (tokA >= 0) ? (x + (size_t)tokA * DDIM) : (const float*)0;
  // B-staging role: thread owns K-octet (8 rows) x 4 n columns, stores packed
  const int bn  = (tid & 31) * 4;
  const int bkq = (tid >> 5) * 8;
  const float* w1p = w1 + ((size_t)e * DDIM + bkq) * HDIM + n0 + bn;

  v8f acc[2][4];
  for (int i = 0; i < 2; i++)
    for (int j = 0; j < 4; j++)
      for (int r = 0; r < 8; r++) acc[i][j][r] = 0.0f;

  for (int k0 = 0; k0 < DDIM; k0 += KT) {
    __syncthreads();
    // stage A (fp32 -> bf16): 32 contiguous K per thread
    __bf16* ap = As + arow * KP + acol;
    if (xrow) {
      const float4* s = (const float4*)(xrow + k0 + acol);
      *(v8bf*)(ap +  0) = cvt8(s[0], s[1]);
      *(v8bf*)(ap +  8) = cvt8(s[2], s[3]);
      *(v8bf*)(ap + 16) = cvt8(s[4], s[5]);
      *(v8bf*)(ap + 24) = cvt8(s[6], s[7]);
    } else {
      v8bf z; for (int i = 0; i < 8; i++) z[i] = (__bf16)0.0f;
      *(v8bf*)(ap +  0) = z; *(v8bf*)(ap +  8) = z;
      *(v8bf*)(ap + 16) = z; *(v8bf*)(ap + 24) = z;
    }
    // stage B transposed: 8 coalesced b128 row loads, register transpose,
    // 4 packed ds_store_b128 along K
    float rv[8][4];
    #pragma unroll
    for (int rr = 0; rr < 8; rr++) {
      float4 tq = *(const float4*)(w1p + (size_t)(k0 + rr) * HDIM);
      rv[rr][0] = tq.x; rv[rr][1] = tq.y; rv[rr][2] = tq.z; rv[rr][3] = tq.w;
    }
    #pragma unroll
    for (int j = 0; j < 4; j++) {
      v8bf o;
      #pragma unroll
      for (int rr = 0; rr < 8; rr++) o[rr] = (__bf16)rv[rr][j];
      *(v8bf*)(Bs + (bn + j) * KP + bkq) = o;
    }
    __syncthreads();

    #pragma unroll
    for (int kk = 0; kk < KT; kk += 32) {
      v16bf af[2], bfm[4];
      #pragma unroll
      for (int ms = 0; ms < 2; ms++) af[ms]  = ld_fragA(As, mblk + ms*16 + lm, half, kk);
      #pragma unroll
      for (int ns = 0; ns < 4; ns++) bfm[ns] = ld_fragB(Bs, nblk + ns*16 + lm, half, kk);
      #pragma unroll
      for (int ms = 0; ms < 2; ms++)
        #pragma unroll
        for (int ns = 0; ns < 4; ns++)
          acc[ms][ns] = __builtin_amdgcn_wmma_f32_16x16x32_bf16(
              false, af[ms], false, bfm[ns], (short)0, acc[ms][ns], false, false);
    }
  }

  #pragma unroll
  for (int ms = 0; ms < 2; ms++)
    for (int ns = 0; ns < 4; ns++) {
      const int n = n0 + nblk + ns*16 + lm;
      const float bias = b1[e * HDIM + n];
      #pragma unroll
      for (int r = 0; r < 8; r++) {
        const int grow = r0 + mblk + ms*16 + half*8 + r;
        float v = acc[ms][ns][r] + bias;
        v = v > 0.0f ? v : 0.0f;
        hws[(size_t)grow * HDIM + n] = (__bf16)v;
      }
    }
}

// ---------------- GEMM2: out[t] += gate * (h @ w2[e] + b2[e]) -------------
__global__ __launch_bounds__(256) void moe_gemm2(
    const __bf16* __restrict__ hws, const float* __restrict__ w2,
    const float* __restrict__ b2, const int* __restrict__ counts,
    const int* __restrict__ offsets, const int* __restrict__ row_token,
    const float* __restrict__ row_gate, float* __restrict__ out)
{
  __shared__ __align__(16) __bf16 As[128 * KP];
  __shared__ __align__(16) __bf16 Bs[128 * KP];
  const int e = blockIdx.z;
  if ((int)blockIdx.y * 128 >= counts[e]) return;
  const int r0 = offsets[e] + blockIdx.y * 128;
  const int n0 = blockIdx.x * 128;

  const int tid  = threadIdx.x;
  const int lane = tid & 31, wave = tid >> 5;
  const int half = lane >> 4, lm = lane & 15;
  const int mblk = (wave & 3) * 32;
  const int nblk = (wave >> 2) * 64;

  const int arow = tid >> 1;
  const int acol = (tid & 1) * 32;
  const __bf16* hrow = hws + (size_t)(r0 + arow) * HDIM + acol;
  const int bn  = (tid & 31) * 4;
  const int bkq = (tid >> 5) * 8;
  const float* w2p = w2 + ((size_t)e * HDIM + bkq) * ODIM + n0 + bn;

  v8f acc[2][4];
  for (int i = 0; i < 2; i++)
    for (int j = 0; j < 4; j++)
      for (int r = 0; r < 8; r++) acc[i][j][r] = 0.0f;

  for (int k0 = 0; k0 < HDIM; k0 += KT) {
    __syncthreads();
    // stage A: already bf16, 32 contiguous K per thread (4x b128)
    __bf16* ap = As + arow * KP + acol;
    *(v8bf*)(ap +  0) = *(const v8bf*)(hrow + k0 +  0);
    *(v8bf*)(ap +  8) = *(const v8bf*)(hrow + k0 +  8);
    *(v8bf*)(ap + 16) = *(const v8bf*)(hrow + k0 + 16);
    *(v8bf*)(ap + 24) = *(const v8bf*)(hrow + k0 + 24);
    // stage B transposed (register transpose, packed b128 stores)
    float rv[8][4];
    #pragma unroll
    for (int rr = 0; rr < 8; rr++) {
      float4 tq = *(const float4*)(w2p + (size_t)(k0 + rr) * ODIM);
      rv[rr][0] = tq.x; rv[rr][1] = tq.y; rv[rr][2] = tq.z; rv[rr][3] = tq.w;
    }
    #pragma unroll
    for (int j = 0; j < 4; j++) {
      v8bf o;
      #pragma unroll
      for (int rr = 0; rr < 8; rr++) o[rr] = (__bf16)rv[rr][j];
      *(v8bf*)(Bs + (bn + j) * KP + bkq) = o;
    }
    __syncthreads();

    #pragma unroll
    for (int kk = 0; kk < KT; kk += 32) {
      v16bf af[2], bfm[4];
      #pragma unroll
      for (int ms = 0; ms < 2; ms++) af[ms]  = ld_fragA(As, mblk + ms*16 + lm, half, kk);
      #pragma unroll
      for (int ns = 0; ns < 4; ns++) bfm[ns] = ld_fragB(Bs, nblk + ns*16 + lm, half, kk);
      #pragma unroll
      for (int ms = 0; ms < 2; ms++)
        #pragma unroll
        for (int ns = 0; ns < 4; ns++)
          acc[ms][ns] = __builtin_amdgcn_wmma_f32_16x16x32_bf16(
              false, af[ms], false, bfm[ns], (short)0, acc[ms][ns], false, false);
    }
  }

  // Scaled scatter-add. Exactly 2 contributions per out element; fp add is
  // commutative, so atomic order does not change the result bitwise.
  #pragma unroll
  for (int ms = 0; ms < 2; ms++)
    #pragma unroll
    for (int r = 0; r < 8; r++) {
      const int grow = r0 + mblk + ms*16 + half*8 + r;
      const int tok = row_token[grow];
      if (tok < 0) continue;
      const float g = row_gate[grow];
      float* orow = out + (size_t)tok * ODIM;
      #pragma unroll
      for (int ns = 0; ns < 4; ns++) {
        const int n = n0 + nblk + ns*16 + lm;
        atomicAdd(orow + n, g * (acc[ms][ns][r] + b2[e * ODIM + n]));
      }
    }
}

extern "C" void kernel_launch(void* const* d_in, const int* in_sizes, int n_in,
                              void* d_out, int out_size, void* d_ws, size_t ws_size,
                              hipStream_t stream) {
  const float* x      = (const float*)d_in[0];
  const float* gate_w = (const float*)d_in[1];
  const float* gate_b = (const float*)d_in[2];
  const float* w1     = (const float*)d_in[3];
  const float* b1     = (const float*)d_in[4];
  const float* w2     = (const float*)d_in[5];
  const float* b2     = (const float*)d_in[6];
  float* out = (float*)d_out;

  char* p = (char*)d_ws;
  int*    counts    = (int*)(p + 0);                       // 8 ints
  int*    offsets   = (int*)(p + 256);                     // 9 ints
  int*    tk_e      = (int*)(p + 512);                     // 2*TOK ints
  int*    tk_p      = (int*)(p + 512 + 65536);             // 2*TOK ints
  float*  tk_v      = (float*)(p + 512 + 131072);          // 2*TOK floats
  int*    row_token = (int*)(p + 197120);                  // ROWS_MAX ints
  float*  row_gate  = (float*)(p + 197120 + ROWS_MAX*4);   // ROWS_MAX floats
  __bf16* hws       = (__bf16*)(p + 336384);               // ROWS_MAX*HDIM bf16

  hipMemsetAsync(counts, 0, NEXP * sizeof(int), stream);
  hipMemsetAsync(row_token, 0xFF, ROWS_MAX * sizeof(int), stream);   // -1
  hipMemsetAsync(out, 0, (size_t)out_size * sizeof(float), stream);

  moe_gate<<<TOK, 256, 0, stream>>>(x, gate_w, gate_b, counts, tk_e, tk_p, tk_v);
  moe_offsets<<<1, 32, 0, stream>>>(counts, offsets);
  moe_scatter<<<(2*TOK)/256, 256, 0, stream>>>(offsets, tk_e, tk_p, tk_v,
                                               row_token, row_gate);
  moe_gemm1<<<dim3(HDIM/128, TOK/128, NEXP), 256, 0, stream>>>(
      x, w1, b1, counts, offsets, row_token, hws);
  moe_gemm2<<<dim3(ODIM/128, TOK/128, NEXP), 256, 0, stream>>>(
      hws, w2, b2, counts, offsets, row_token, row_gate, out);
}
